// SPAIRGlimpseRGBDecoder_64269890617425
// MI455X (gfx1250) — matrix operand order repaired and checked by reading, not compiled, not run on hardware
//
#include <hip/hip_runtime.h>
#include <hip/hip_bf16.h>
#include <math.h>

// ---------------------------------------------------------------------------
// CDNA5 (gfx1250) fused PointNetConv stack using v_wmma_f32_16x16x32_bf16.
// ---------------------------------------------------------------------------

typedef __attribute__((ext_vector_type(16))) __bf16 v16bf;
typedef __attribute__((ext_vector_type(8)))  float  v8f;

union Frag16 {
    v16bf   v;
    unsigned u[8];
};

__device__ __forceinline__ unsigned short f2bf(float f) {
    unsigned u = __float_as_uint(f);
    unsigned r = u + 0x7FFFu + ((u >> 16) & 1u);   // round-to-nearest-even
    return (unsigned short)(r >> 16);
}

// ---------------------------------------------------------------------------
// Pre-pack a row-major weight matrix W[Krows x Ncols] into per-lane WMMA
// B-fragment layout for 16x16x32 bf16:
//   out[((kb*NB+nb)*32 + lane)*8 + v] = pack(W[k][col], W[k+1][col])
//   k   = kb*32 + 2*(v&3) + (lane>=16 ? 8 : 0) + (v>=4 ? 16 : 0)
//   col = nb*16 + (lane & 15)
// Rows >= Krows are zero-padded (used when c_mid=16 < K tile of 32).
// ---------------------------------------------------------------------------
__global__ void pack_b_kernel(const float* __restrict__ W, int Krows, int Ncols,
                              int KB, int NB, unsigned* __restrict__ outp)
{
    int t = blockIdx.x * blockDim.x + threadIdx.x;
    int total = KB * NB * 32 * 8;
    if (t >= total) return;
    int v     = t & 7;
    int lane  = (t >> 3) & 31;
    int tilei = t >> 8;
    int nb    = tilei % NB;
    int kb    = tilei / NB;
    int col   = nb * 16 + (lane & 15);
    int k     = kb * 32 + 2 * (v & 3) + ((lane >= 16) ? 8 : 0) + ((v >= 4) ? 16 : 0);
    float f0 = (k     < Krows) ? W[k * Ncols + col]       : 0.0f;
    float f1 = (k + 1 < Krows) ? W[(k + 1) * Ncols + col] : 0.0f;
    outp[t] = (unsigned)f2bf(f0) | ((unsigned)f2bf(f1) << 16);
}

__global__ void fill_zero_kernel(unsigned* __restrict__ p, int n)
{
    int i = blockIdx.x * blockDim.x + threadIdx.x;
    if (i < n) p[i] = 0u;
}

// ---------------------------------------------------------------------------
// Fused edge kernel: gather + bf16 WMMA GEMM + rel/bias tail + ReLU +
// segment-max (uint atomicMax on non-negative float bits).
//   msg = relu( [x[src], pos[src]-pos[dst]] @ Wl + bl )     (16 edges / wave)
//   agg[dst] = max(agg[dst], msg)
// ---------------------------------------------------------------------------
template<int C_IN, int C_MID>
__global__ __launch_bounds__(128)
void edge_conv_kernel(const float* __restrict__ x,     // N x C_IN
                      const float* __restrict__ pos,   // N x 3
                      const int*   __restrict__ ei,    // 2 x E
                      const unsigned* __restrict__ Bpre,
                      const float* __restrict__ Wl,    // (C_IN+3) x C_MID
                      const float* __restrict__ bl,    // C_MID
                      unsigned*    __restrict__ agg,   // N x C_MID (f32 bits)
                      int nE)
{
    constexpr int WV   = 4;
    constexpr int ASTR = C_IN + 2;          // padded bf16 row stride (odd dwords)
    constexpr int KB   = C_IN / 32;
    constexpr int NB   = C_MID / 16;

    const int lane = threadIdx.x & 31;
    const int wave = threadIdx.x >> 5;
    const int tile = blockIdx.x * WV + wave;
    const int e0   = tile * 16;

    __shared__ unsigned short Ast[WV][16 * ASTR];
    __shared__ float relS[WV][16][3];
    __shared__ int   srcS[WV][16];
    __shared__ int   dstS[WV][16];
    __shared__ float WtS[3][C_MID];
    __shared__ float blS[C_MID];

    // Phase 1: stage weight tail rows + bias (block) and edge meta (wave).
    for (int i = threadIdx.x; i < C_MID; i += blockDim.x) {
        WtS[0][i] = Wl[(C_IN + 0) * C_MID + i];
        WtS[1][i] = Wl[(C_IN + 1) * C_MID + i];
        WtS[2][i] = Wl[(C_IN + 2) * C_MID + i];
        blS[i]    = bl[i];
    }
    if (lane < 16) {
        int e = min(e0 + lane, nE - 1);     // clamp: duplicate edge is a no-op for max
        int s = ei[e];
        int d = ei[nE + e];
        srcS[wave][lane] = s;
        dstS[wave][lane] = d;
        relS[wave][lane][0] = pos[s * 3 + 0] - pos[d * 3 + 0];
        relS[wave][lane][1] = pos[s * 3 + 1] - pos[d * 3 + 1];
        relS[wave][lane][2] = pos[s * 3 + 2] - pos[d * 3 + 2];
    }
    __syncthreads();

    // Phase 2: gather x[src] rows -> bf16 LDS tile (16 x C_IN).
    for (int i = lane; i < 16 * C_IN; i += 32) {
        int row = i / C_IN;
        int col = i - row * C_IN;
        int s   = srcS[wave][row];
        Ast[wave][row * ASTR + col] = f2bf(x[s * C_IN + col]);
    }
    __syncthreads();

    const int rowA  = lane & 15;
    const int rbase = (lane >= 16) ? 8 : 0;
    const int khalf = (lane >= 16) ? 8 : 0;

    for (int nb = 0; nb < NB; ++nb) {
        v8f acc = {0.f, 0.f, 0.f, 0.f, 0.f, 0.f, 0.f, 0.f};
#pragma unroll
        for (int kb = 0; kb < KB; ++kb) {
            Frag16 a, b;
#pragma unroll
            for (int v = 0; v < 8; ++v) {
                int k0 = kb * 32 + ((v & 3) << 1) + khalf + ((v >= 4) ? 16 : 0);
                a.u[v] = *(const unsigned*)&Ast[wave][rowA * ASTR + k0];
            }
            const uint4* bp = (const uint4*)(Bpre + (((kb * NB + nb) * 32 + lane) << 3));
            uint4 b0 = bp[0], b1 = bp[1];
            b.u[0] = b0.x; b.u[1] = b0.y; b.u[2] = b0.z; b.u[3] = b0.w;
            b.u[4] = b1.x; b.u[5] = b1.y; b.u[6] = b1.z; b.u[7] = b1.w;
            acc = __builtin_amdgcn_wmma_f32_16x16x32_bf16(
                false, a.v, false, b.v, (short)0, acc, false, false);
        }
        const int col = nb * 16 + (lane & 15);
        const float w0 = WtS[0][col], w1 = WtS[1][col], w2 = WtS[2][col];
        const float bb = blS[col];
#pragma unroll
        for (int m = 0; m < 8; ++m) {
            int row = m + rbase;
            float v = acc[m]
                    + relS[wave][row][0] * w0
                    + relS[wave][row][1] * w1
                    + relS[wave][row][2] * w2 + bb;
            v = fmaxf(v, 0.0f);               // ReLU -> non-negative
            if (v > 0.0f) {
                int d = dstS[wave][row];
                atomicMax(&agg[d * C_MID + col], __float_as_uint(v));
            }
        }
    }
}

// ---------------------------------------------------------------------------
// Node kernel: out = celu(agg @ Wg + bg), 16 rows per wave, WMMA bf16.
// K=16 case zero-pads to a single K=32 tile (Bpre is pre-padded too).
// ---------------------------------------------------------------------------
template<int K, int C_OUT>
__global__ __launch_bounds__(128)
void node_gemm_celu_kernel(const float* __restrict__ agg,   // N x K
                           const unsigned* __restrict__ Bpre,
                           const float* __restrict__ bg,    // C_OUT
                           float* __restrict__ out,         // N x C_OUT
                           int nRows)
{
    constexpr int WV   = 4;
    constexpr int KP   = (K < 32) ? 32 : K;
    constexpr int ASTR = KP + 2;
    constexpr int KB   = KP / 32;
    constexpr int NB   = C_OUT / 16;

    const int lane = threadIdx.x & 31;
    const int wave = threadIdx.x >> 5;
    const int tile = blockIdx.x * WV + wave;
    const int r0   = tile * 16;

    __shared__ unsigned short Ast[WV][16 * ASTR];
    __shared__ float bgS[C_OUT];

    for (int i = threadIdx.x; i < C_OUT; i += blockDim.x) bgS[i] = bg[i];

    for (int i = lane; i < 16 * KP; i += 32) {
        int row = i / KP;
        int col = i - row * KP;
        int r   = min(r0 + row, nRows - 1);
        float f = (col < K) ? agg[r * K + col] : 0.0f;
        Ast[wave][row * ASTR + col] = f2bf(f);
    }
    __syncthreads();

    const int rowA  = lane & 15;
    const int rbase = (lane >= 16) ? 8 : 0;
    const int khalf = (lane >= 16) ? 8 : 0;

    for (int nb = 0; nb < NB; ++nb) {
        v8f acc = {0.f, 0.f, 0.f, 0.f, 0.f, 0.f, 0.f, 0.f};
#pragma unroll
        for (int kb = 0; kb < KB; ++kb) {
            Frag16 a, b;
#pragma unroll
            for (int v = 0; v < 8; ++v) {
                int k0 = kb * 32 + ((v & 3) << 1) + khalf + ((v >= 4) ? 16 : 0);
                a.u[v] = *(const unsigned*)&Ast[wave][rowA * ASTR + k0];
            }
            const uint4* bp = (const uint4*)(Bpre + (((kb * NB + nb) * 32 + lane) << 3));
            uint4 b0 = bp[0], b1 = bp[1];
            b.u[0] = b0.x; b.u[1] = b0.y; b.u[2] = b0.z; b.u[3] = b0.w;
            b.u[4] = b1.x; b.u[5] = b1.y; b.u[6] = b1.z; b.u[7] = b1.w;
            acc = __builtin_amdgcn_wmma_f32_16x16x32_bf16(
                false, a.v, false, b.v, (short)0, acc, false, false);
        }
        const int col = nb * 16 + (lane & 15);
        const float bb = bgS[col];
#pragma unroll
        for (int m = 0; m < 8; ++m) {
            int r = r0 + m + rbase;
            if (r < nRows) {
                float v = acc[m] + bb;
                v = (v > 0.0f) ? v : (expf(v) - 1.0f);   // CELU(alpha=1)
                out[r * C_OUT + col] = v;
            }
        }
    }
}

// ---------------------------------------------------------------------------
// Final Linear(16 -> 3): one thread per node row.
// ---------------------------------------------------------------------------
__global__ void final_linear_kernel(const float* __restrict__ x,   // N x 16
                                    const float* __restrict__ Wo,  // 16 x 3
                                    const float* __restrict__ bo,  // 3
                                    float* __restrict__ out, int n)
{
    int i = blockIdx.x * blockDim.x + threadIdx.x;
    if (i >= n) return;
    float r0 = bo[0], r1 = bo[1], r2 = bo[2];
#pragma unroll
    for (int k = 0; k < 16; ++k) {
        float xv = x[i * 16 + k];
        r0 = fmaf(xv, Wo[k * 3 + 0], r0);
        r1 = fmaf(xv, Wo[k * 3 + 1], r1);
        r2 = fmaf(xv, Wo[k * 3 + 2], r2);
    }
    out[i * 3 + 0] = r0;
    out[i * 3 + 1] = r1;
    out[i * 3 + 2] = r2;
}

// ---------------------------------------------------------------------------
extern "C" void kernel_launch(void* const* d_in, const int* in_sizes, int n_in,
                              void* d_out, int out_size, void* d_ws, size_t ws_size,
                              hipStream_t stream)
{
    const float* z_what = (const float*)d_in[0];
    const float* pos1   = (const float*)d_in[1];
    const float* pos2   = (const float*)d_in[2];
    const float* pos3   = (const float*)d_in[3];
    const int*   e1     = (const int*)d_in[4];
    const int*   e2     = (const int*)d_in[5];
    const int*   e3     = (const int*)d_in[6];
    const float* W1l = (const float*)d_in[7];  const float* b1l = (const float*)d_in[8];
    const float* W1g = (const float*)d_in[9];  const float* b1g = (const float*)d_in[10];
    const float* W2l = (const float*)d_in[11]; const float* b2l = (const float*)d_in[12];
    const float* W2g = (const float*)d_in[13]; const float* b2g = (const float*)d_in[14];
    const float* W3l = (const float*)d_in[15]; const float* b3l = (const float*)d_in[16];
    const float* W3g = (const float*)d_in[17]; const float* b3g = (const float*)d_in[18];
    const float* Wo  = (const float*)d_in[19]; const float* bo  = (const float*)d_in[20];

    const int N = in_sizes[0] / 128;
    const int E = in_sizes[4] / 2;

    // ---- workspace layout (floats) ----
    float* ws = (float*)d_ws;
    size_t o = 0;
    unsigned* agg1 = (unsigned*)(ws + o); o += (size_t)N * 128;
    float*    x1   = ws + o;              o += (size_t)N * 64;
    unsigned* agg2 = (unsigned*)(ws + o); o += (size_t)N * 32;
    float*    x2   = ws + o;              o += (size_t)N * 32;
    unsigned* agg3 = (unsigned*)(ws + o); o += (size_t)N * 16;
    float*    x3   = ws + o;              o += (size_t)N * 16;
    unsigned* P    = (unsigned*)(ws + o);
    unsigned* P1l = P;            // KB=4 NB=8 -> 8192 u32
    unsigned* P1g = P1l + 8192;   // KB=4 NB=4 -> 4096
    unsigned* P2l = P1g + 4096;   // KB=2 NB=2 -> 1024
    unsigned* P2g = P2l + 1024;   // KB=1 NB=2 -> 512
    unsigned* P3l = P2g + 512;    // KB=1 NB=1 -> 256
    unsigned* P3g = P3l + 256;    // KB=1 NB=1 -> 256 (K=16 zero-padded to 32)

    // ---- pre-pack weights into WMMA B-fragment layout ----
    pack_b_kernel<<<(8192 + 255) / 256, 256, 0, stream>>>(W1l, 128, 128, 4, 8, P1l);
    pack_b_kernel<<<(4096 + 255) / 256, 256, 0, stream>>>(W1g, 128,  64, 4, 4, P1g);
    pack_b_kernel<<<(1024 + 255) / 256, 256, 0, stream>>>(W2l,  64,  32, 2, 2, P2l);
    pack_b_kernel<<<( 512 + 255) / 256, 256, 0, stream>>>(W2g,  32,  32, 1, 2, P2g);
    pack_b_kernel<<<( 256 + 255) / 256, 256, 0, stream>>>(W3l,  32,  16, 1, 1, P3l);
    pack_b_kernel<<<( 256 + 255) / 256, 256, 0, stream>>>(W3g,  16,  16, 1, 1, P3g);

    const int eBlocks = ((E + 15) / 16 + 3) / 4;   // 16 edges/wave, 4 waves/block
    const int nBlocks = ((N + 15) / 16 + 3) / 4;   // 16 rows/wave, 4 waves/block

    // ---- layer 1: 128 -> 128 -> 64 ----
    { int n = N * 128; fill_zero_kernel<<<(n + 255) / 256, 256, 0, stream>>>(agg1, n); }
    edge_conv_kernel<128, 128><<<eBlocks, 128, 0, stream>>>(
        z_what, pos1, e1, P1l, W1l, b1l, agg1, E);
    node_gemm_celu_kernel<128, 64><<<nBlocks, 128, 0, stream>>>(
        (const float*)agg1, P1g, b1g, x1, N);

    // ---- layer 2: 64 -> 32 -> 32 ----
    { int n = N * 32; fill_zero_kernel<<<(n + 255) / 256, 256, 0, stream>>>(agg2, n); }
    edge_conv_kernel<64, 32><<<eBlocks, 128, 0, stream>>>(
        x1, pos2, e2, P2l, W2l, b2l, agg2, E);
    node_gemm_celu_kernel<32, 32><<<nBlocks, 128, 0, stream>>>(
        (const float*)agg2, P2g, b2g, x2, N);

    // ---- layer 3: 32 -> 16 -> 16 ----
    { int n = N * 16; fill_zero_kernel<<<(n + 255) / 256, 256, 0, stream>>>(agg3, n); }
    edge_conv_kernel<32, 16><<<eBlocks, 128, 0, stream>>>(
        x2, pos3, e3, P3l, W3l, b3l, agg3, E);
    node_gemm_celu_kernel<16, 16><<<nBlocks, 128, 0, stream>>>(
        (const float*)agg3, P3g, b3g, x3, N);

    // ---- final Linear(16 -> 3) ----
    final_linear_kernel<<<(N + 255) / 256, 256, 0, stream>>>(x3, Wo, bo, (float*)d_out, N);
}
